// PhysicsNetwork_v2_34591666602339
// MI455X (gfx1250) — matrix operand order repaired and checked by reading, not compiled, not run on hardware
//
#include <hip/hip_runtime.h>

// ---------------------------------------------------------------------------
// MI455X (gfx1250) fused physics-network forward.
// Compute-bound (~95 GFLOP, <100MB traffic) -> bf16 WMMA 16x16x32, f32 accum.
// All GEMM tiles: 16 rows per block (8 wave32s), A from LDS, B (weights)
// pre-transposed to [OUT][K] bf16 so each lane loads its B column contiguously.
// Pairwise chain fully fused per scene: no P=262144-row intermediates.
// ELU uses branchless v_exp_f32 (abs err ~6e-8 vs expm1, < bf16 ulp).
// ---------------------------------------------------------------------------

typedef __bf16 v16bf __attribute__((ext_vector_type(16)));
typedef __bf16 v8bf  __attribute__((ext_vector_type(8)));
typedef float  v8f   __attribute__((ext_vector_type(8)));

#define NTOT 32768   // bs*K
#define NSCENE 4096  // bs

__device__ __forceinline__ float eluf(float x) {
  float e = __expf(x) - 1.f;        // branchless: v_exp_f32 + add + cndmask
  return x > 0.f ? x : e;
}
__device__ __forceinline__ float sigmf(float x) { return 1.f / (1.f + __expf(-x)); }

// A-fragment (16x32 bf16): lane = row M (lane&15); K chunks {half*16..+7} and
// {half*16+8..+15} (contiguous, matches B).  rowPtr points at LDS row, c1 = kt*32+half*16.
__device__ __forceinline__ v16bf load_a_frag(const __bf16* rowPtr, int c1) {
  v8bf lo = *(const v8bf*)(rowPtr + c1);
  v8bf hi = *(const v8bf*)(rowPtr + c1 + 8);
  return __builtin_shufflevector(lo, hi, 0,1,2,3,4,5,6,7,8,9,10,11,12,13,14,15);
}

// One 16x16 output tile, full K reduction.  Wt is [OUT][KDIM] bf16 (transposed
// weights): B-fragment lane = col N (lane&15), 16 consecutive K per lane,
// half-wave offset +16 in K (this code's contiguous k order; A uses the same).
template<int KDIM>
__device__ __forceinline__ v8f gemm16(const __bf16* As, int aStride,
                                      const __bf16* Wt, int n0,
                                      float binit, int lane) {
  v8f acc;
  #pragma unroll
  for (int q = 0; q < 8; ++q) acc[q] = binit;
  const int m    = lane & 15;
  const int half = lane >> 4;
  const __bf16* arow = As + m * aStride;
  const __bf16* bcol = Wt + (size_t)(n0 + m) * KDIM + half * 16;
  #pragma unroll
  for (int kt = 0; kt < KDIM / 32; ++kt) {
    v16bf a = load_a_frag(arow, kt * 32 + half * 16);
    v16bf b = *(const v16bf*)(bcol + kt * 32);
    acc = __builtin_amdgcn_wmma_f32_16x16x32_bf16(false, a, false, b,
                                                  (short)0, acc, false, false);
  }
  return acc;
}

// D layout: lane = N (lane&15), VGPR r -> M = r + 8*half.  elu + store to LDS.
__device__ __forceinline__ void store_act_lds(v8f acc, __bf16* Hs, int hStride,
                                              int n0, int lane) {
  const int N = lane & 15, half = lane >> 4;
  #pragma unroll
  for (int r = 0; r < 8; ++r)
    Hs[(r + 8 * half) * hStride + n0 + N] = (__bf16)eluf(acc[r]);
}

// ---------------------------------------------------------------------------
// Generic fused 2-layer MLP: Y = act2(elu(X@W1+b1)@W2+b2)   (16 rows / block)
// XM: 0 = f32 input width SRCW zero-padded to IN; 1 = concat of two bf16[*,128];
//     2 = bf16 input width IN.
// ---------------------------------------------------------------------------
template<int IN, int SRCW, int HID, int OUT, int XM, bool ELU2, bool SCALE>
__global__ __launch_bounds__(256) void mlp2_kernel(
    const float* __restrict__ Xf,
    const __bf16* __restrict__ Xa, const __bf16* __restrict__ Xb,
    const __bf16* __restrict__ W1t, const float* __restrict__ b1,
    const __bf16* __restrict__ W2t, const float* __restrict__ b2,
    const float* __restrict__ rowscale,
    __bf16* __restrict__ Ybf, float* __restrict__ Yf) {
  __shared__ __align__(16) __bf16 Xs[16 * (IN + 8)];
  __shared__ __align__(16) __bf16 Hs[16 * (HID + 8)];
  const int row0 = blockIdx.x * 16;
  const int tid  = threadIdx.x;

  for (int idx = tid; idx < 16 * IN; idx += 256) {
    const int r = idx / IN, c = idx - r * IN;
    float v;
    if (XM == 0)      v = (c < SRCW) ? Xf[(size_t)(row0 + r) * SRCW + c] : 0.f;
    else if (XM == 1) v = (c < 128) ? (float)Xa[(size_t)(row0 + r) * 128 + c]
                                    : (float)Xb[(size_t)(row0 + r) * 128 + (c - 128)];
    else              v = (float)Xa[(size_t)(row0 + r) * IN + c];
    Xs[r * (IN + 8) + c] = (__bf16)v;
  }
  __syncthreads();

  const int wave = tid >> 5, lane = tid & 31;
  for (int ct = wave; ct < HID / 16; ct += 8) {
    const int n0 = ct * 16;
    v8f acc = gemm16<IN>(Xs, IN + 8, W1t, n0, b1[n0 + (lane & 15)], lane);
    store_act_lds(acc, Hs, HID + 8, n0, lane);
  }
  __syncthreads();

  for (int ct = wave; ct < OUT / 16; ct += 8) {
    const int n0 = ct * 16;
    v8f acc = gemm16<HID>(Hs, HID + 8, W2t, n0, b2[n0 + (lane & 15)], lane);
    const int N = lane & 15, half = lane >> 4;
    #pragma unroll
    for (int r = 0; r < 8; ++r) {
      const int row = row0 + r + 8 * half;
      float v = acc[r];
      if (ELU2)  v = eluf(v);
      if (SCALE) v *= rowscale[row];
      if (Ybf) Ybf[(size_t)row * OUT + n0 + N] = (__bf16)v;
      if (Yf)  Yf [(size_t)row * OUT + n0 + N] = v;
    }
  }
}

// act_attn: elu(concat(s,ae)@W1+b1) then [128]->1 dot + sigmoid -> att[row]
__global__ __launch_bounds__(256) void act_attn_kernel(
    const __bf16* __restrict__ Xa, const __bf16* __restrict__ Xb,
    const __bf16* __restrict__ W1t, const float* __restrict__ b1,
    const float* __restrict__ w2, const float* __restrict__ b2,
    float* __restrict__ att) {
  __shared__ __align__(16) __bf16 Xs[16 * 264];
  __shared__ __align__(16) __bf16 Hs[16 * 136];
  const int row0 = blockIdx.x * 16, tid = threadIdx.x;
  for (int idx = tid; idx < 16 * 256; idx += 256) {
    const int r = idx >> 8, c = idx & 255;
    float v = (c < 128) ? (float)Xa[(size_t)(row0 + r) * 128 + c]
                        : (float)Xb[(size_t)(row0 + r) * 128 + (c - 128)];
    Xs[r * 264 + c] = (__bf16)v;
  }
  __syncthreads();
  const int wave = tid >> 5, lane = tid & 31;
  {
    const int n0 = wave * 16;
    v8f acc = gemm16<256>(Xs, 264, W1t, n0, b1[n0 + (lane & 15)], lane);
    store_act_lds(acc, Hs, 136, n0, lane);
  }
  __syncthreads();
  if (tid < 16) {
    float s = b2[0];
    for (int c = 0; c < 128; ++c) s += (float)Hs[tid * 136 + c] * w2[c];
    att[row0 + tid] = sigmf(s);
  }
}

// ---------------------------------------------------------------------------
// Fused pairwise chain, one block per scene (K=8 -> 64 pairs = 4 row-tiles).
// pf(i,j)=[enc_i|enc_j] -> pairwise(256,256,128) -> int_attn -> int_eff ->
// tot[i] = sum_{j!=i} elu(eff)*att.  Within a row-tile, i = 2t+half (constant
// per lane-half) and j = r (accumulator index), so the reduction is a per-lane
// register sum and each tot[i][col] is written exactly once. No atomics.
// ---------------------------------------------------------------------------
__global__ __launch_bounds__(256) void pairwise_fused_kernel(
    const __bf16* __restrict__ enc,
    const __bf16* __restrict__ Wp1t, const float* __restrict__ bp1,
    const __bf16* __restrict__ Wp2t, const float* __restrict__ bp2,
    const __bf16* __restrict__ We1t, const float* __restrict__ be1,
    const __bf16* __restrict__ We2t, const float* __restrict__ be2,
    const __bf16* __restrict__ Wa1t, const float* __restrict__ ba1,
    const float* __restrict__ wa2, const float* __restrict__ ba2,
    __bf16* __restrict__ tot) {
  __shared__ __align__(16) __bf16 encS[8 * 136];
  __shared__ __align__(16) __bf16 H1[16 * 264];
  __shared__ __align__(16) __bf16 PW[16 * 136];
  __shared__ __align__(16) __bf16 H2[16 * 136];
  __shared__ float PATT[16];
  const int b = blockIdx.x, tid = threadIdx.x;
  {
    const int i = tid >> 5, c = (tid & 31) * 4;
    const __bf16* src = enc + ((size_t)b * 8 + i) * 128 + c;
    __bf16* dst = encS + i * 136 + c;
    dst[0] = src[0]; dst[1] = src[1]; dst[2] = src[2]; dst[3] = src[3];
  }
  __syncthreads();
  const int wave = tid >> 5, lane = tid & 31, m = lane & 15, half = lane >> 4;

  for (int t = 0; t < 4; ++t) {
    const int im = 2 * t + (m >> 3), jm = m & 7;  // A-row m -> pair (im, jm)
    const __bf16* ri = encS + im * 136;
    const __bf16* rj = encS + jm * 136;

    // pairwise L1: 256 -> 256 (A gathered from concat [enc_i | enc_j])
    for (int ct = wave; ct < 16; ct += 8) {
      const int n0 = ct * 16;
      const float bv = bp1[n0 + m];
      v8f acc;
      #pragma unroll
      for (int q = 0; q < 8; ++q) acc[q] = bv;
      const __bf16* bcol = Wp1t + (size_t)(n0 + m) * 256 + half * 16;
      #pragma unroll
      for (int kt = 0; kt < 8; ++kt) {
        const int c1 = kt * 32 + half * 16;
        const int c2 = c1 + 8;
        v8bf lo = *(const v8bf*)((c1 < 128) ? (ri + c1) : (rj + (c1 - 128)));
        v8bf hi = *(const v8bf*)((c2 < 128) ? (ri + c2) : (rj + (c2 - 128)));
        v16bf a = __builtin_shufflevector(lo, hi, 0,1,2,3,4,5,6,7,8,9,10,11,12,13,14,15);
        v16bf bb = *(const v16bf*)(bcol + kt * 32);
        acc = __builtin_amdgcn_wmma_f32_16x16x32_bf16(false, a, false, bb,
                                                      (short)0, acc, false, false);
      }
      store_act_lds(acc, H1, 264, n0, lane);
    }
    __syncthreads();
    { // pairwise L2: 256 -> 128
      const int n0 = wave * 16;
      v8f acc = gemm16<256>(H1, 264, Wp2t, n0, bp2[n0 + m], lane);
      store_act_lds(acc, PW, 136, n0, lane);
    }
    __syncthreads();
    { // int_attn L1: 128 -> 128
      const int n0 = wave * 16;
      v8f acc = gemm16<128>(PW, 136, Wa1t, n0, ba1[n0 + m], lane);
      store_act_lds(acc, H2, 136, n0, lane);
    }
    __syncthreads();
    if (tid < 16) { // int_attn L2: 128 -> 1, sigmoid
      float s = ba2[0];
      for (int c = 0; c < 128; ++c) s += (float)H2[tid * 136 + c] * wa2[c];
      PATT[tid] = sigmf(s);
    }
    __syncthreads();
    { // int_effect L1: 128 -> 128 (reuse H2)
      const int n0 = wave * 16;
      v8f acc = gemm16<128>(PW, 136, We1t, n0, be1[n0 + m], lane);
      store_act_lds(acc, H2, 136, n0, lane);
    }
    __syncthreads();
    { // int_effect L2 + masked attention-weighted reduction over j
      const int n0 = wave * 16;
      v8f acc = gemm16<128>(H2, 136, We2t, n0, be2[n0 + m], lane);
      const int iskip = 2 * t + half;  // this half's entity i (== j to skip)
      float s = 0.f;
      #pragma unroll
      for (int r = 0; r < 8; ++r)
        if (r != iskip) s += eluf(acc[r]) * PATT[r + 8 * half];
      tot[((size_t)b * 8 + iskip) * 128 + n0 + m] = (__bf16)s;
    }
    __syncthreads();
  }
}

// Weight convert: src f32 [IN_src][OUT] row-major -> dst bf16 [OUT][IN_dst]
// (transposed, K zero-padded) matching the WMMA B-fragment layout.
__global__ void convt_kernel(const float* __restrict__ src, __bf16* __restrict__ dst,
                             int IN_src, int IN_dst, int OUT) {
  const int idx = blockIdx.x * 256 + threadIdx.x;
  if (idx >= OUT * IN_dst) return;
  const int o = idx / IN_dst, i = idx - o * IN_dst;
  dst[idx] = (__bf16)((i < IN_src) ? src[(size_t)i * OUT + o] : 0.f);
}

// ---------------------------------------------------------------------------
extern "C" void kernel_launch(void* const* d_in, const int* in_sizes, int n_in,
                              void* d_out, int out_size, void* d_ws, size_t ws_size,
                              hipStream_t stream) {
  (void)out_size; (void)ws_size;
  // canonical ids: 0 inertia, 1 action_enc, 2 act_effect, 3 act_attn,
  // 4 pairwise, 5 int_effect, 6 int_attn, 7 final, 8 lam1, 9 lam2
  struct M { const float *W1, *b1, *W2, *b2; };
  M mp[10];
  int ssIdx, actIdx, pstart; bool sortedNest;
  if (n_in >= 3 && in_sizes[0] == 1) {          // fully tree-sorted flatten
    ssIdx = n_in - 1; actIdx = 1; pstart = 2; sortedNest = true;
  } else {                                       // top-level dict order
    ssIdx = 0; actIdx = 1; pstart = 2;
    sortedNest = (in_sizes[2] == 32768);        // act_attn.W1 first if sorted
  }
  static const int sortedMap[10] = {3, 2, 1, 7, 0, 6, 5, 8, 9, 4};
  for (int s = 0; s < 10; ++s) {
    const int cid = sortedNest ? sortedMap[s] : s;
    const float* a0 = (const float*)d_in[pstart + s * 4 + 0];
    const float* a1 = (const float*)d_in[pstart + s * 4 + 1];
    const float* a2 = (const float*)d_in[pstart + s * 4 + 2];
    const float* a3 = (const float*)d_in[pstart + s * 4 + 3];
    if (sortedNest) { mp[cid].W1 = a0; mp[cid].W2 = a1; mp[cid].b1 = a2; mp[cid].b2 = a3; }
    else            { mp[cid].W1 = a0; mp[cid].b1 = a1; mp[cid].W2 = a2; mp[cid].b2 = a3; }
  }
  const float* ss   = (const float*)d_in[ssIdx];
  const float* acts = (const float*)d_in[actIdx];

  // ---- workspace carve-out (~26 MB) ----
  char* wsb = (char*)d_ws; size_t off = 0;
  auto alloc  = [&](size_t bytes) -> void* {
    off = (off + 255) & ~(size_t)255;
    void* p = wsb + off; off += bytes; return p;
  };
  auto allocB = [&](size_t elems) { return (__bf16*)alloc(elems * 2); };

  __bf16* wi1  = allocB(128 * 128);  __bf16* wi2  = allocB(128 * 128);
  __bf16* wac1 = allocB(128 * 32);   __bf16* wac2 = allocB(128 * 128);
  __bf16* wae1 = allocB(128 * 256);  __bf16* wae2 = allocB(128 * 128);
  __bf16* wat1 = allocB(128 * 256);
  __bf16* wp1  = allocB(256 * 256);  __bf16* wp2  = allocB(128 * 256);
  __bf16* wie1 = allocB(128 * 128);  __bf16* wie2 = allocB(128 * 128);
  __bf16* wia1 = allocB(128 * 128);
  __bf16* wf1  = allocB(128 * 256);  __bf16* wf2  = allocB(128 * 128);
  __bf16* wl11 = allocB(128 * 128);  __bf16* wl12 = allocB(128 * 128);
  __bf16* wl21 = allocB(128 * 128);  __bf16* wl22 = allocB(128 * 128);
  __bf16* sbf  = allocB((size_t)NTOT * 128);   // state_enc_flat; later reused as new_state
  __bf16* aebf = allocB((size_t)NTOT * 128);   // action_enc;     later reused as total_effect
  __bf16* encb = allocB((size_t)NTOT * 128);   // state_enc (eff*att)
  float*  attf = (float*)alloc((size_t)NTOT * 4);
  __bf16* totb = aebf;   // ae dead after act_effect
  __bf16* nsb  = sbf;    // s dead after pairwise/final inputs staged

  auto convt = [&](const float* src, __bf16* dst, int is, int id, int o) {
    const int tot = o * id;
    convt_kernel<<<(tot + 255) / 256, 256, 0, stream>>>(src, dst, is, id, o);
  };
  convt(mp[0].W1, wi1, 128, 128, 128);  convt(mp[0].W2, wi2, 128, 128, 128);
  convt(mp[1].W1, wac1,   4,  32, 128); convt(mp[1].W2, wac2, 128, 128, 128);
  convt(mp[2].W1, wae1, 256, 256, 128); convt(mp[2].W2, wae2, 128, 128, 128);
  convt(mp[3].W1, wat1, 256, 256, 128);
  convt(mp[4].W1, wp1,  256, 256, 256); convt(mp[4].W2, wp2,  256, 256, 128);
  convt(mp[5].W1, wie1, 128, 128, 128); convt(mp[5].W2, wie2, 128, 128, 128);
  convt(mp[6].W1, wia1, 128, 128, 128);
  convt(mp[7].W1, wf1,  256, 256, 128); convt(mp[7].W2, wf2,  128, 128, 128);
  convt(mp[8].W1, wl11, 128, 128, 128); convt(mp[8].W2, wl12, 128, 128, 128);
  convt(mp[9].W1, wl21, 128, 128, 128); convt(mp[9].W2, wl22, 128, 128, 128);

  const dim3 grid(NTOT / 16), blk(256);

  // inertia: elu(elu(x@W1+b1)@W2+b2)
  mlp2_kernel<128, 128, 128, 128, 0, true, false><<<grid, blk, 0, stream>>>(
      ss, nullptr, nullptr, wi1, mp[0].b1, wi2, mp[0].b2, nullptr, sbf, nullptr);
  // action_enc: in=4 zero-padded to K=32
  mlp2_kernel<32, 4, 128, 128, 0, true, false><<<grid, blk, 0, stream>>>(
      acts, nullptr, nullptr, wac1, mp[1].b1, wac2, mp[1].b2, nullptr, aebf, nullptr);
  // act_attn: sigmoid head -> att[N]
  act_attn_kernel<<<grid, blk, 0, stream>>>(
      sbf, aebf, wat1, mp[3].b1, mp[3].W2, mp[3].b2, attf);
  // act_effect scaled by att -> state_enc
  mlp2_kernel<256, 256, 128, 128, 1, true, true><<<grid, blk, 0, stream>>>(
      nullptr, sbf, aebf, wae1, mp[2].b1, wae2, mp[2].b2, attf, encb, nullptr);
  // fused pairwise chain -> total_effect
  pairwise_fused_kernel<<<dim3(NSCENE), blk, 0, stream>>>(
      encb, wp1, mp[4].b1, wp2, mp[4].b2,
      wie1, mp[5].b1, wie2, mp[5].b2,
      wia1, mp[6].b1, mp[6].W2, mp[6].b2, totb);
  // final: concat(enc, tot) -> new_state
  mlp2_kernel<256, 256, 128, 128, 1, true, false><<<grid, blk, 0, stream>>>(
      nullptr, encb, totb, wf1, mp[7].b1, wf2, mp[7].b2, nullptr, nsb, nullptr);
  // lam1 / lam2: identity output act, f32 straight to d_out
  float* out = (float*)d_out;
  mlp2_kernel<128, 128, 128, 128, 2, false, false><<<grid, blk, 0, stream>>>(
      nullptr, nsb, nullptr, wl11, mp[8].b1, wl12, mp[8].b2, nullptr, nullptr, out);
  mlp2_kernel<128, 128, 128, 128, 2, false, false><<<grid, blk, 0, stream>>>(
      nullptr, nsb, nullptr, wl21, mp[9].b1, wl22, mp[9].b2, nullptr, nullptr,
      out + (size_t)NTOT * 128);
}
